// PointTransformerEncoderSmall_9165460210298
// MI455X (gfx1250) — matrix operand order repaired and checked by reading, not compiled, not run on hardware
//
#include <hip/hip_runtime.h>

// ---------------------------------------------------------------------------
// PCT-style point transformer encoder for MI455X (gfx1250, wave32).
// GEMM stages (local_op grouped convolutions) use v_wmma_f32_16x16x32_f16 with
// the B (weight) tile brought into LDS via GLOBAL_LOAD_ASYNC_TO_LDS_B128
// (ASYNCcnt), overlapping the A-tile gather/compute.  Everything else is
// latency-bound (13MB working set << 192MB L2) -> lean VALU kernels with
// register-resident selection loops.
// ---------------------------------------------------------------------------

#define BB   32      // batch
#define NPT  16384   // points
#define S1   128     // fps1 centroids
#define NB1  32      // knn1 neighbors
#define S2   32      // fps2 centroids
#define NB2  16      // knn2 neighbors
#define CCH  64      // feature channels
#define DD   128     // grouped row width (2*CCH)

static __device__ __forceinline__ float bn_rs() { return rsqrtf(1.0f + 1e-5f); }

typedef _Float16 v16h __attribute__((ext_vector_type(16)));
typedef float    v8f  __attribute__((ext_vector_type(8)));

// Load a 16-half WMMA fragment as two 16B LDS reads (both 16B aligned).
static __device__ __forceinline__ v16h ld_frag(const _Float16* p0, const _Float16* p1) {
    union U { uint4 u[2]; v16h h; } r;
    r.u[0] = *reinterpret_cast<const uint4*>(p0);
    r.u[1] = *reinterpret_cast<const uint4*>(p1);
    return r.h;
}

// Async copy of 16 bytes global -> LDS, tracked by ASYNCcnt (CDNA5 §15.18.3).
// GVS mode: lds_addr VGPR, 32-bit byte-offset VGPR, 64-bit SGPR base.
static __device__ __forceinline__ void async_g2l_b128(unsigned lds_addr,
                                                      unsigned gl_off,
                                                      const void* gbase) {
    asm volatile("global_load_async_to_lds_b128 %0, %1, %2 offset:0"
                 :: "v"(lds_addr), "v"(gl_off), "s"(gbase) : "memory");
}
static __device__ __forceinline__ void wait_asynccnt0() {
    asm volatile("s_wait_asynccnt 0" ::: "memory");
}
// LDS byte address of a shared-memory pointer (low 32 bits of the pointer).
static __device__ __forceinline__ unsigned lds_addr_of(const void* p) {
    return (unsigned)(uintptr_t)p;
}

// conv1 (6->64) + fake-bn + relu for a single channel
static __device__ __forceinline__ float conv1_feat(const float* w6, float g, float bb,
                                                   float x0, float x1, float x2,
                                                   float x3, float x4, float x5) {
    float a = w6[0]*x0 + w6[1]*x1 + w6[2]*x2 + w6[3]*x3 + w6[4]*x4 + w6[5]*x5;
    return fmaxf(a * (g * bn_rs()) + bb, 0.0f);
}

// ---------------------------------------------------------------------------
// K0: one-shot f32->f16 conversion of the two local_op weight matrices so the
// 4096 GEMM blocks can async-DMA the 16KB f16 tile instead of each converting
// 32KB of f32 (saves ~33M redundant v_cvt and halves weight L2 traffic).
// ---------------------------------------------------------------------------
__global__ __launch_bounds__(256) void k_prep(const float* __restrict__ l0_w,
                                              const float* __restrict__ l1_w,
                                              _Float16* __restrict__ w0h,
                                              _Float16* __restrict__ w1h) {
    const int i = blockIdx.x * 256 + threadIdx.x;   // grid 32 -> 8192 elems
    w0h[i] = (_Float16)l0_w[i];
    w1h[i] = (_Float16)l1_w[i];
}

// ---------------------------------------------------------------------------
// K1: center = mean(xyz, axis=1); writes d_out[0:96] and ws.center
// ---------------------------------------------------------------------------
__global__ __launch_bounds__(256) void k_center(const float* __restrict__ xyz,
                                                float* __restrict__ center_ws,
                                                float* __restrict__ out) {
    __shared__ float sx[256], sy[256], sz[256];
    const int b = blockIdx.x, t = threadIdx.x;
    float ax = 0.f, ay = 0.f, az = 0.f;
    for (int i = t; i < NPT; i += 256) {
        const float* p = xyz + ((size_t)b * NPT + i) * 3;
        ax += p[0]; ay += p[1]; az += p[2];
    }
    sx[t] = ax; sy[t] = ay; sz[t] = az; __syncthreads();
    for (int off = 128; off > 0; off >>= 1) {
        if (t < off) { sx[t] += sx[t+off]; sy[t] += sy[t+off]; sz[t] += sz[t+off]; }
        __syncthreads();
    }
    if (t == 0) {
        const float inv = 1.0f / (float)NPT;
        const float c0 = sx[0]*inv, c1 = sy[0]*inv, c2 = sz[0]*inv;
        center_ws[b*3+0] = c0; center_ws[b*3+1] = c1; center_ws[b*3+2] = c2;
        out[b*3+0] = c0; out[b*3+1] = c1; out[b*3+2] = c2;
    }
}

// ---------------------------------------------------------------------------
// K2: farthest point sampling, 128 of 16384.  dist kept in registers
// (16/thread @ 1024 threads); argmax via LDS tree, tie -> lowest index.
// ---------------------------------------------------------------------------
__global__ __launch_bounds__(1024) void k_fps1(const float* __restrict__ xyz,
                                               const float* __restrict__ center,
                                               int* __restrict__ fps,
                                               float* __restrict__ nxyz) {
    __shared__ float rv[1024];
    __shared__ int   ri[1024];
    const int b = blockIdx.x, t = threadIdx.x;
    const float cx = center[b*3+0], cy = center[b*3+1], cz = center[b*3+2];
    float px[16], py[16], pz[16], dd[16];
#pragma unroll
    for (int i = 0; i < 16; ++i) {
        const int idx = t + i * 1024;
        const float* p = xyz + ((size_t)b * NPT + idx) * 3;
        px[i] = p[0]-cx; py[i] = p[1]-cy; pz[i] = p[2]-cz;
        dd[i] = 1e10f;
    }
    int cur = 0;
    if (t == 0) {
        fps[b*S1] = 0;
        const float* p = xyz + (size_t)b * NPT * 3;
        nxyz[(b*S1)*3+0] = p[0]-cx; nxyz[(b*S1)*3+1] = p[1]-cy; nxyz[(b*S1)*3+2] = p[2]-cz;
    }
#pragma unroll 1
    for (int it = 1; it < S1; ++it) {
        const float* cp = xyz + ((size_t)b * NPT + cur) * 3;
        const float gx = cp[0]-cx, gy = cp[1]-cy, gz = cp[2]-cz;
        float bv = -1.0f; int bi = 0;
#pragma unroll
        for (int i = 0; i < 16; ++i) {
            const float dx = px[i]-gx, dy = py[i]-gy, dz = pz[i]-gz;
            const float d = dx*dx + dy*dy + dz*dz;
            dd[i] = fminf(dd[i], d);
            if (dd[i] > bv) { bv = dd[i]; bi = t + i * 1024; }
        }
        rv[t] = bv; ri[t] = bi; __syncthreads();
        for (int off = 512; off > 0; off >>= 1) {
            if (t < off) {
                const float ov = rv[t+off]; const int oi = ri[t+off];
                if (ov > rv[t] || (ov == rv[t] && oi < ri[t])) { rv[t] = ov; ri[t] = oi; }
            }
            __syncthreads();
        }
        const int w = ri[0];
        __syncthreads();
        if (t == 0) {
            fps[b*S1+it] = w;
            const float* wp = xyz + ((size_t)b * NPT + w) * 3;
            nxyz[(b*S1+it)*3+0] = wp[0]-cx;
            nxyz[(b*S1+it)*3+1] = wp[1]-cy;
            nxyz[(b*S1+it)*3+2] = wp[2]-cz;
        }
        cur = w;
    }
}

// ---------------------------------------------------------------------------
// K3: 32-NN of each centroid among 16384 pts. 64 dists/thread in registers,
// 32 lexicographic-min selections via LDS tree.
// ---------------------------------------------------------------------------
__global__ __launch_bounds__(256) void k_knn1(const float* __restrict__ xyz,
                                              const float* __restrict__ center,
                                              const float* __restrict__ nxyz,
                                              int* __restrict__ idx1) {
    __shared__ float rv[256];
    __shared__ int   ri[256];
    __shared__ float pdv;
    __shared__ int   pdi;
    const int s = blockIdx.x, b = blockIdx.y, t = threadIdx.x;
    const float cx = center[b*3+0], cy = center[b*3+1], cz = center[b*3+2];
    const float qx = nxyz[(b*S1+s)*3+0], qy = nxyz[(b*S1+s)*3+1], qz = nxyz[(b*S1+s)*3+2];
    float dc[64];
#pragma unroll
    for (int u = 0; u < 64; ++u) {
        const int i = t + u * 256;
        const float* p = xyz + ((size_t)b * NPT + i) * 3;
        if ((u & 7) == 0) __builtin_prefetch(p + 8 * 256 * 3, 0, 1);
        const float dx = (p[0]-cx)-qx, dy = (p[1]-cy)-qy, dz = (p[2]-cz)-qz;
        dc[u] = dx*dx + dy*dy + dz*dz;
    }
    float pd = -1e30f; int pi = -1;
#pragma unroll 1
    for (int k = 0; k < NB1; ++k) {
        float bv = 3e38f; int bi = 0x7fffffff;
#pragma unroll
        for (int u = 0; u < 64; ++u) {
            const int i = t + u * 256;
            const float d = dc[u];
            const bool ok = (d > pd) || (d == pd && i > pi);
            if (ok && (d < bv || (d == bv && i < bi))) { bv = d; bi = i; }
        }
        rv[t] = bv; ri[t] = bi; __syncthreads();
        for (int off = 128; off > 0; off >>= 1) {
            if (t < off) {
                const float ov = rv[t+off]; const int oi = ri[t+off];
                if (ov < rv[t] || (ov == rv[t] && oi < ri[t])) { rv[t] = ov; ri[t] = oi; }
            }
            __syncthreads();
        }
        if (t == 0) { idx1[((size_t)(b*S1+s))*NB1 + k] = ri[0]; pdv = rv[0]; pdi = ri[0]; }
        __syncthreads();
        pd = pdv; pi = pdi;
        __syncthreads();
    }
}

// ---------------------------------------------------------------------------
// K4: grouped GEMM #1 via WMMA.  One block per (b,s):
//  - async-DMA the 16KB f16 weight tile into LDS (ASYNCcnt) at kernel entry,
//  - meanwhile recompute conv1+bn+relu features for 32 neighbors + centroid,
//  - 8 waves x 4x v_wmma_f32_16x16x32_f16 (M=32,N=64,K=128),
//  - fused bn+relu+max-over-k epilogue (int atomicMax valid since relu>=0).
// ---------------------------------------------------------------------------
__global__ __launch_bounds__(256) void k_group1(
    const float* __restrict__ xyz, const float* __restrict__ f,
    const float* __restrict__ center, const float* __restrict__ conv1_w,
    const float* __restrict__ bn1_g, const float* __restrict__ bn1_b,
    const int* __restrict__ fps, const float* __restrict__ nxyz,
    const int* __restrict__ idx1, const _Float16* __restrict__ w0h,
    const float* __restrict__ l0_g, const float* __restrict__ l0_b,
    float* __restrict__ feat0) {

    __shared__ __attribute__((aligned(16))) _Float16 Alds[NB1 * DD]; // 32x128
    __shared__ __attribute__((aligned(16))) _Float16 Blds[CCH * DD]; // 64x128 (W row-major)
    __shared__ float cenf[CCH];
    __shared__ int   colmax[CCH];

    const int s = blockIdx.x, b = blockIdx.y, t = threadIdx.x;
    const float cx = center[b*3+0], cy = center[b*3+1], cz = center[b*3+2];

    // Kick off the async B-tile copy first: 4 x 256 lanes x 16B = 16KB.
    {
        const unsigned lbase = lds_addr_of(Blds);
#pragma unroll
        for (int i = 0; i < 4; ++i) {
            const unsigned off = (unsigned)(t + i * 256) * 16u;
            async_g2l_b128(lbase + off, off, (const void*)w0h);
        }
    }

    if (t < CCH) colmax[t] = 0;
    if (t < CCH) {
        const int ci = fps[b*S1 + s];
        const float* q  = nxyz + (b*S1 + s) * 3;          // centered centroid xyz
        const float* qf = f + ((size_t)b * NPT + ci) * 3; // its raw f features
        cenf[t] = conv1_feat(conv1_w + t*6, bn1_g[t], bn1_b[t],
                             q[0], q[1], q[2], qf[0], qf[1], qf[2]);
    }
    __syncthreads();

    { // build A rows: [nbr_feat - cen_feat | cen_feat]
        const int row = t >> 3, c8 = t & 7;
        const int j = idx1[((size_t)(b*S1+s))*NB1 + row];
        const float* p  = xyz + ((size_t)b * NPT + j) * 3;
        const float* pf = f   + ((size_t)b * NPT + j) * 3;
        const float x0 = p[0]-cx, x1 = p[1]-cy, x2 = p[2]-cz;
#pragma unroll
        for (int i = 0; i < 8; ++i) {
            const int c = c8*8 + i;
            const float fc = conv1_feat(conv1_w + c*6, bn1_g[c], bn1_b[c],
                                        x0, x1, x2, pf[0], pf[1], pf[2]);
            Alds[row*DD + c]       = (_Float16)(fc - cenf[c]);
            Alds[row*DD + CCH + c] = (_Float16)cenf[c];
        }
    }
    wait_asynccnt0();          // our wave's B-tile chunks have landed in LDS
    __syncthreads();           // everyone's chunks + A tile visible

    const int w = t >> 5, lane = t & 31;
    const int mt = w & 1, nt = w >> 1;          // 2 row tiles x 4 col tiles
    const int l15 = lane & 15, hi = lane >> 4;
    v8f acc = {};
#pragma unroll
    for (int kk = 0; kk < 4; ++kk) {
        // A layout (ISA 7.12.2, 16-bit A 16x32): lanes<16 K[0..7|16..23], lanes>=16 +8
        const _Float16* ap = Alds + (mt*16 + l15) * DD + kk*32 + hi*8;
        const v16h af = ld_frag(ap, ap + 16);
        // B layout: lane=n, same K order as A: K[0..7|16..23], lanes>=16 +8
        const _Float16* bp = Blds + (nt*16 + l15) * DD + kk*32 + hi*8;
        const v16h bf = ld_frag(bp, bp + 16);
        acc = __builtin_amdgcn_wmma_f32_16x16x32_f16(false, af, false, bf,
                                                     (short)0, acc, false, false);
    }
    const int o = nt*16 + l15;
    const float sc = l0_g[o] * bn_rs(), bi = l0_b[o];
    float m = 0.0f;
#pragma unroll
    for (int r = 0; r < 8; ++r) m = fmaxf(m, fmaxf(acc[r]*sc + bi, 0.0f));
    atomicMax(&colmax[o], __float_as_int(m));
    __syncthreads();
    if (t < CCH) feat0[((size_t)(b*S1+s))*CCH + t] = __int_as_float(colmax[t]);
}

// ---------------------------------------------------------------------------
// K5: FPS (32 of 128) + 16-NN among the 128 centroids. One block per batch.
// ---------------------------------------------------------------------------
__global__ __launch_bounds__(128) void k_fps2knn2(const float* __restrict__ nxyz,
                                                  int* __restrict__ fps2,
                                                  int* __restrict__ idx2) {
    __shared__ float rv[128];
    __shared__ int   ri[128];
    __shared__ float cenc[3];
    __shared__ int   selbuf[S2];
    __shared__ float pdv;
    __shared__ int   pdi;
    const int b = blockIdx.x, t = threadIdx.x;
    const float* p = nxyz + (b*S1 + t) * 3;
    const float px = p[0], py = p[1], pz = p[2];
    float dist = 1e10f;
    int cur = 0;
    if (t == 0) { fps2[b*S2] = 0; selbuf[0] = 0; }
    for (int it = 1; it < S2; ++it) {
        if (t == cur) { cenc[0] = px; cenc[1] = py; cenc[2] = pz; }
        __syncthreads();
        const float dx = px-cenc[0], dy = py-cenc[1], dz = pz-cenc[2];
        dist = fminf(dist, dx*dx + dy*dy + dz*dz);
        rv[t] = dist; ri[t] = t; __syncthreads();
        for (int off = 64; off > 0; off >>= 1) {
            if (t < off) {
                const float ov = rv[t+off]; const int oi = ri[t+off];
                if (ov > rv[t] || (ov == rv[t] && oi < ri[t])) { rv[t] = ov; ri[t] = oi; }
            }
            __syncthreads();
        }
        cur = ri[0];
        __syncthreads();
        if (t == 0) { fps2[b*S2+it] = cur; selbuf[it] = cur; }
        __syncthreads();
    }
    for (int s2 = 0; s2 < S2; ++s2) {
        const int cidx = selbuf[s2];
        if (t == cidx) { cenc[0] = px; cenc[1] = py; cenc[2] = pz; }
        __syncthreads();
        const float dx = px-cenc[0], dy = py-cenc[1], dz = pz-cenc[2];
        const float d = dx*dx + dy*dy + dz*dz;
        float pd = -1e30f; int pi = -1;
        for (int k = 0; k < NB2; ++k) {
            const bool ok = (d > pd) || (d == pd && t > pi);
            rv[t] = ok ? d : 3e38f;
            ri[t] = ok ? t : 0x7fffffff;
            __syncthreads();
            for (int off = 64; off > 0; off >>= 1) {
                if (t < off) {
                    const float ov = rv[t+off]; const int oi = ri[t+off];
                    if (ov < rv[t] || (ov == rv[t] && oi < ri[t])) { rv[t] = ov; ri[t] = oi; }
                }
                __syncthreads();
            }
            if (t == 0) { idx2[(b*S2+s2)*NB2 + k] = ri[0]; pdv = rv[0]; pdi = ri[0]; }
            __syncthreads();
            pd = pdv; pi = pdi;
            __syncthreads();
        }
    }
}

// ---------------------------------------------------------------------------
// K6: grouped GEMM #2 via WMMA (M=16,N=64,K=128 per block; 4 waves), same
// async B-tile scheme.  Output feat1 stored (b, n=32, c=64).
// ---------------------------------------------------------------------------
__global__ __launch_bounds__(128) void k_group2(
    const float* __restrict__ feat0, const int* __restrict__ fps2,
    const int* __restrict__ idx2, const _Float16* __restrict__ w1h,
    const float* __restrict__ l1_g, const float* __restrict__ l1_b,
    float* __restrict__ feat1) {

    __shared__ __attribute__((aligned(16))) _Float16 Alds[NB2 * DD]; // 16x128
    __shared__ __attribute__((aligned(16))) _Float16 Blds[CCH * DD]; // 64x128
    __shared__ float cenf[CCH];
    __shared__ int   colmax[CCH];

    const int s = blockIdx.x, b = blockIdx.y, t = threadIdx.x;

    { // async B-tile copy: 8 x 128 lanes x 16B = 16KB
        const unsigned lbase = lds_addr_of(Blds);
#pragma unroll
        for (int i = 0; i < 8; ++i) {
            const unsigned off = (unsigned)(t + i * 128) * 16u;
            async_g2l_b128(lbase + off, off, (const void*)w1h);
        }
    }

    if (t < CCH) {
        colmax[t] = 0;
        const int ci = fps2[b*S2 + s];
        cenf[t] = feat0[((size_t)(b*S1 + ci))*CCH + t];
    }
    __syncthreads();

    {
        const int row = t >> 3, c8 = t & 7;          // 16 rows x 8 threads
        const int j = idx2[(b*S2+s)*NB2 + row];
        const float* pf = feat0 + ((size_t)(b*S1 + j)) * CCH;
#pragma unroll
        for (int i = 0; i < 8; ++i) {
            const int c = c8*8 + i;
            const float fc = pf[c];
            Alds[row*DD + c]       = (_Float16)(fc - cenf[c]);
            Alds[row*DD + CCH + c] = (_Float16)cenf[c];
        }
    }
    wait_asynccnt0();
    __syncthreads();

    const int w = t >> 5, lane = t & 31;
    const int nt = w;                              // 4 col tiles, single row tile
    const int l15 = lane & 15, hi = lane >> 4;
    v8f acc = {};
#pragma unroll
    for (int kk = 0; kk < 4; ++kk) {
        const _Float16* ap = Alds + l15 * DD + kk*32 + hi*8;
        const v16h af = ld_frag(ap, ap + 16);
        const _Float16* bp = Blds + (nt*16 + l15) * DD + kk*32 + hi*8;
        const v16h bf = ld_frag(bp, bp + 16);
        acc = __builtin_amdgcn_wmma_f32_16x16x32_f16(false, af, false, bf,
                                                     (short)0, acc, false, false);
    }
    const int o = nt*16 + l15;
    const float sc = l1_g[o] * bn_rs(), bi = l1_b[o];
    float m = 0.0f;
#pragma unroll
    for (int r = 0; r < 8; ++r) m = fmaxf(m, fmaxf(acc[r]*sc + bi, 0.0f));
    atomicMax(&colmax[o], __float_as_int(m));
    __syncthreads();
    if (t < CCH) feat1[((size_t)(b*S2+s))*CCH + t] = __int_as_float(colmax[t]);
}

// ---------------------------------------------------------------------------
// K7: per-batch tail: st-convs, 2x offset-attention, fuse, global max, MLP.
// < 4 MFLOP/batch, latency-bound -> VALU in LDS (~48KB).
// ---------------------------------------------------------------------------
__device__ void sa_layer_dev(int t, float (*Xin)[32], float (*Xout)[32],
                             float (*XV)[32], float (*XR)[32],
                             float (*q)[16], float (*att)[32],
                             const float* qk_w, const float* v_w, const float* v_b,
                             const float* t_w, const float* t_b,
                             const float* g, const float* bb) {
    for (int i = t; i < 32*16; i += 256) {
        const int n = i >> 4, d = i & 15;
        float a = 0.f;
        for (int c = 0; c < 64; ++c) a += qk_w[d*64+c] * Xin[c][n];
        q[n][d] = a;
    }
    __syncthreads();
    for (int i = t; i < 32*32; i += 256) {
        const int n = i >> 5, m = i & 31;
        float a = 0.f;
#pragma unroll
        for (int d = 0; d < 16; ++d) a += q[n][d] * q[m][d];
        att[n][m] = a;
    }
    __syncthreads();
    if (t < 32) { // softmax over last axis (rows)
        float mx = -3e38f;
        for (int m = 0; m < 32; ++m) mx = fmaxf(mx, att[t][m]);
        float sm = 0.f;
        for (int m = 0; m < 32; ++m) { const float e = __expf(att[t][m]-mx); att[t][m] = e; sm += e; }
        const float inv = 1.0f / sm;
        for (int m = 0; m < 32; ++m) att[t][m] *= inv;
    }
    __syncthreads();
    if (t < 32) { // column renorm (axis=1 sum)
        float s = 0.f;
        for (int n = 0; n < 32; ++n) s += att[n][t];
        const float inv = 1.0f / (1e-9f + s);
        for (int n = 0; n < 32; ++n) att[n][t] *= inv;
    }
    __syncthreads();
    for (int i = t; i < 64*32; i += 256) { // xv
        const int o = i >> 5, n = i & 31;
        float a = v_b[o];
        for (int c = 0; c < 64; ++c) a += v_w[o*64+c] * Xin[c][n];
        XV[o][n] = a;
    }
    __syncthreads();
    for (int i = t; i < 64*32; i += 256) { // x_r = xv @ attn
        const int c = i >> 5, m = i & 31;
        float a = 0.f;
        for (int n = 0; n < 32; ++n) a += XV[c][n] * att[n][m];
        XR[c][m] = a;
    }
    __syncthreads();
    for (int i = t; i < 64*32; i += 256) { // trans conv + bn + relu + residual
        const int o = i >> 5, n = i & 31;
        float a = t_b[o];
        for (int c = 0; c < 64; ++c) a += t_w[o*64+c] * (Xin[c][n] - XR[c][n]);
        const float y = fmaxf(a * (g[o] * bn_rs()) + bb[o], 0.0f);
        Xout[o][n] = Xin[o][n] + y;
    }
    __syncthreads();
}

__global__ __launch_bounds__(256) void k_tail(
    const float* __restrict__ feat1,
    const float* __restrict__ st1w, const float* __restrict__ st1g, const float* __restrict__ st1b,
    const float* __restrict__ st2w, const float* __restrict__ st2g, const float* __restrict__ st2b,
    const float* __restrict__ qk1, const float* __restrict__ vw1, const float* __restrict__ vb1,
    const float* __restrict__ tw1, const float* __restrict__ tb1,
    const float* __restrict__ g1,  const float* __restrict__ b1,
    const float* __restrict__ qk2, const float* __restrict__ vw2, const float* __restrict__ vb2,
    const float* __restrict__ tw2, const float* __restrict__ tb2,
    const float* __restrict__ g2,  const float* __restrict__ b2,
    const float* __restrict__ fuse_w, const float* __restrict__ fuse_g, const float* __restrict__ fuse_b,
    const float* __restrict__ lin1_w, const float* __restrict__ bn6_g, const float* __restrict__ bn6_b,
    const float* __restrict__ lin2_w, const float* __restrict__ lin2_b,
    float* __restrict__ out) {

    __shared__ float X0[64][32], Xa[64][32], Xb[64][32], Xc[64][32], Xd[64][32];
    __shared__ float q[32][16], att[32][32];
    __shared__ float gvec[256], hvec[256];

    const int b = blockIdx.x, t = threadIdx.x;
    for (int i = t; i < 64*32; i += 256) {
        const int n = i & 31, c = i >> 5;
        X0[c][n] = feat1[((size_t)(b*S2 + n))*CCH + c];   // feature_1 (c,n)
    }
    __syncthreads();
    for (int i = t; i < 64*32; i += 256) { // st conv1
        const int o = i >> 5, n = i & 31;
        float a = 0.f;
        for (int c = 0; c < 64; ++c) a += st1w[o*64+c] * X0[c][n];
        Xa[o][n] = fmaxf(a * (st1g[o]*bn_rs()) + st1b[o], 0.0f);
    }
    __syncthreads();
    for (int i = t; i < 64*32; i += 256) { // st conv2
        const int o = i >> 5, n = i & 31;
        float a = 0.f;
        for (int c = 0; c < 64; ++c) a += st2w[o*64+c] * Xa[c][n];
        Xb[o][n] = fmaxf(a * (st2g[o]*bn_rs()) + st2b[o], 0.0f);
    }
    __syncthreads();
    sa_layer_dev(t, Xb, Xa, Xc, Xd, q, att, qk1, vw1, vb1, tw1, tb1, g1, b1); // x1 -> Xa
    sa_layer_dev(t, Xa, Xb, Xc, Xd, q, att, qk2, vw2, vb2, tw2, tb2, g2, b2); // x2 -> Xb
    { // fuse (192->256), leaky-relu(0.2), max over n
        const int o = t;
        float mx = -3e38f;
        for (int n = 0; n < 32; ++n) {
            float a = 0.f;
            for (int c = 0; c < 64; ++c) a += fuse_w[o*192 + c]       * Xa[c][n];
            for (int c = 0; c < 64; ++c) a += fuse_w[o*192 + 64 + c]  * Xb[c][n];
            for (int c = 0; c < 64; ++c) a += fuse_w[o*192 + 128 + c] * X0[c][n];
            float y = a * (fuse_g[o]*bn_rs()) + fuse_b[o];
            y = (y > 0.f) ? y : 0.2f * y;
            mx = fmaxf(mx, y);
        }
        gvec[o] = mx;
    }
    __syncthreads();
    { // lin1 + bn6 + relu
        float a = 0.f;
        for (int c = 0; c < 256; ++c) a += lin1_w[t*256+c] * gvec[c];
        hvec[t] = fmaxf(a * (bn6_g[t]*bn_rs()) + bn6_b[t], 0.0f);
    }
    __syncthreads();
    { // lin2 + bias
        float a = lin2_b[t];
        for (int c = 0; c < 256; ++c) a += lin2_w[t*256+c] * hvec[c];
        out[96 + b*256 + t] = a;
    }
}

// ---------------------------------------------------------------------------
// launch
// ---------------------------------------------------------------------------
extern "C" void kernel_launch(void* const* d_in, const int* in_sizes, int n_in,
                              void* d_out, int out_size, void* d_ws, size_t ws_size,
                              hipStream_t stream) {
    const float* xyz     = (const float*)d_in[0];
    const float* f       = (const float*)d_in[1];
    const float* conv1_w = (const float*)d_in[2];
    const float* bn1_g   = (const float*)d_in[3];
    const float* bn1_b   = (const float*)d_in[4];
    const float* l0_w    = (const float*)d_in[5];
    const float* l0_g    = (const float*)d_in[6];
    const float* l0_b    = (const float*)d_in[7];
    const float* l1_w    = (const float*)d_in[8];
    const float* l1_g    = (const float*)d_in[9];
    const float* l1_b    = (const float*)d_in[10];
    const float* st_c1_w = (const float*)d_in[11];
    const float* st_bn1g = (const float*)d_in[12];
    const float* st_bn1b = (const float*)d_in[13];
    const float* st_c2_w = (const float*)d_in[14];
    const float* st_bn2g = (const float*)d_in[15];
    const float* st_bn2b = (const float*)d_in[16];
    const float* sa1_qk  = (const float*)d_in[17];
    const float* sa1_vw  = (const float*)d_in[18];
    const float* sa1_vb  = (const float*)d_in[19];
    const float* sa1_tw  = (const float*)d_in[20];
    const float* sa1_tb  = (const float*)d_in[21];
    const float* sa1_g   = (const float*)d_in[22];
    const float* sa1_b   = (const float*)d_in[23];
    const float* sa2_qk  = (const float*)d_in[24];
    const float* sa2_vw  = (const float*)d_in[25];
    const float* sa2_vb  = (const float*)d_in[26];
    const float* sa2_tw  = (const float*)d_in[27];
    const float* sa2_tb  = (const float*)d_in[28];
    const float* sa2_g   = (const float*)d_in[29];
    const float* sa2_b   = (const float*)d_in[30];
    const float* fuse_w  = (const float*)d_in[31];
    const float* fuse_g  = (const float*)d_in[32];
    const float* fuse_b  = (const float*)d_in[33];
    const float* lin1_w  = (const float*)d_in[34];
    const float* bn6_g   = (const float*)d_in[35];
    const float* bn6_b   = (const float*)d_in[36];
    const float* lin2_w  = (const float*)d_in[37];
    const float* lin2_b  = (const float*)d_in[38];
    float* out = (float*)d_out;

    // workspace layout (~2MB total, all 16B aligned)
    char* w = (char*)d_ws;
    size_t off = 0;
    float*    center = (float*)   (w + off); off += 512;
    int*      fps1   = (int*)     (w + off); off += (size_t)BB*S1*sizeof(int);
    float*    nxyz   = (float*)   (w + off); off += (size_t)BB*S1*3*sizeof(float);
    int*      idx1   = (int*)     (w + off); off += (size_t)BB*S1*NB1*sizeof(int);
    float*    feat0  = (float*)   (w + off); off += (size_t)BB*S1*CCH*sizeof(float);
    int*      fps2   = (int*)     (w + off); off += (size_t)BB*S2*sizeof(int);
    int*      idx2   = (int*)     (w + off); off += (size_t)BB*S2*NB2*sizeof(int);
    float*    feat1  = (float*)   (w + off); off += (size_t)BB*S2*CCH*sizeof(float);
    _Float16* w0h    = (_Float16*)(w + off); off += (size_t)CCH*DD*sizeof(_Float16);
    _Float16* w1h    = (_Float16*)(w + off); off += (size_t)CCH*DD*sizeof(_Float16);
    (void)off; (void)ws_size; (void)n_in; (void)in_sizes; (void)out_size;

    k_prep  <<<BB, 256, 0, stream>>>(l0_w, l1_w, w0h, w1h);
    k_center<<<BB, 256, 0, stream>>>(xyz, center, out);
    k_fps1  <<<BB, 1024, 0, stream>>>(xyz, center, fps1, nxyz);
    k_knn1  <<<dim3(S1, BB), 256, 0, stream>>>(xyz, center, nxyz, idx1);
    k_group1<<<dim3(S1, BB), 256, 0, stream>>>(xyz, f, center, conv1_w, bn1_g, bn1_b,
                                               fps1, nxyz, idx1, w0h, l0_g, l0_b, feat0);
    k_fps2knn2<<<BB, 128, 0, stream>>>(nxyz, fps2, idx2);
    k_group2<<<dim3(S2, BB), 128, 0, stream>>>(feat0, fps2, idx2, w1h, l1_g, l1_b, feat1);
    k_tail  <<<BB, 256, 0, stream>>>(feat1,
                                     st_c1_w, st_bn1g, st_bn1b,
                                     st_c2_w, st_bn2g, st_bn2b,
                                     sa1_qk, sa1_vw, sa1_vb, sa1_tw, sa1_tb, sa1_g, sa1_b,
                                     sa2_qk, sa2_vw, sa2_vb, sa2_tw, sa2_tb, sa2_g, sa2_b,
                                     fuse_w, fuse_g, fuse_b,
                                     lin1_w, bn6_g, bn6_b, lin2_w, lin2_b,
                                     out);
}
